// GNN_91242285236265
// MI455X (gfx1250) — matrix operand order repaired and checked by reading, compiled, not run on hardware
//
#include <hip/hip_runtime.h>
#include <hip/hip_bf16.h>

// ---------------------------------------------------------------------------
// Problem constants (from the reference)
// ---------------------------------------------------------------------------
#define NN    50000
#define EE    800000
#define F_IN  512
#define HH    256
#define G4    1024          // 4*H LSTM gate width

typedef __attribute__((ext_vector_type(16))) _Float16 v16h;
typedef __attribute__((ext_vector_type(8)))  float    v8f;

#define TILE_M 128
#define TILE_N 128
#define TILE_K 32
#define LDA_S  40   // As row stride in halfs (32 + pad)

enum { GF_RELU = 1, GF_ACC = 2 };

union FragH { v16h v; uint4 q[2]; };

// ---------------------------------------------------------------------------
// WMMA GEMM: C[M,Nc] = op(A_f16[M,K] @ B + bias*bias_scale [+C])
// A row-major [M,K] (staged through LDS).
// BT is the weight matrix pre-TRANSPOSED: BT[n*K + k]  (so each lane's WMMA
// B-fragment is two 8-half runs (k 8h+0..7, 16+8h+0..7) -> two global_load_b128 from L2).
// Nc multiple of 128, K multiple of 32.
// ---------------------------------------------------------------------------
__global__ __launch_bounds__(256)
void wmma_gemm(const _Float16* __restrict__ A, const _Float16* __restrict__ BT,
               const float* __restrict__ bias, float bias_scale,
               float* __restrict__ C, _Float16* __restrict__ Ch,
               int M, int K, int Nc, int flags)
{
    __shared__ _Float16 As[TILE_M * LDA_S];

    const int tid    = threadIdx.x;
    const int wave   = tid >> 5;
    const int lane   = tid & 31;
    const int half16 = (lane >> 4) & 1;   // lane group 0-15 vs 16-31
    const int l16    = lane & 15;
    const int wm     = wave >> 1;         // 0..3 : wave row (32 rows each)
    const int wn     = wave & 1;          // 0..1 : wave col (64 cols each)
    const int m0     = blockIdx.y * TILE_M;
    const int n0     = blockIdx.x * TILE_N;

    // global->LDS staging assignment for A: 128 rows x 2 half-rows of 16 halfs
    const int arow = tid >> 1, aseg = tid & 1;
    const _Float16* agp = A + (size_t)(m0 + arow) * K + aseg * 16;
    const bool arow_ok = (m0 + arow) < M;

    v8f acc[2][4] = {};

    for (int k0 = 0; k0 < K; k0 += TILE_K) {
        __syncthreads();
        // ---- stage A tile (guard M edge) ----
        {
            uint4 va0 = {}, va1 = {};
            if (arow_ok) {
                const uint4* p = (const uint4*)(agp + k0);
                va0 = p[0]; va1 = p[1];
                if (k0 + TILE_K < K)                      // prefetch next A tile
                    __builtin_prefetch(agp + k0 + TILE_K, 0, 0);
            }
            uint4* q = (uint4*)(&As[arow * LDA_S + aseg * 16]);
            q[0] = va0; q[1] = va1;
        }
        __syncthreads();

        // ---- A fragments from LDS (ISA 7.12.2 16-bit A layout) ----
        FragH af[2];
        #pragma unroll
        for (int mi = 0; mi < 2; ++mi) {
            const _Float16* ap = &As[(wm * 32 + mi * 16 + l16) * LDA_S + half16 * 8];
            af[mi].q[0] = *(const uint4*)(ap);        // K = 0..7   (+half*8)
            af[mi].q[1] = *(const uint4*)(ap + 16);   // K = 16..23 (+half*8)
        }
        // ---- B fragments straight from transposed weights in L2 ----
        FragH bf[4];
        #pragma unroll
        for (int ni = 0; ni < 4; ++ni) {
            const int n = n0 + wn * 64 + ni * 16 + l16;
            const _Float16* bp = BT + (size_t)n * K + k0 + half16 * 8;
            bf[ni].q[0] = *(const uint4*)(bp);
            bf[ni].q[1] = *(const uint4*)(bp + 16);
        }

        #pragma unroll
        for (int mi = 0; mi < 2; ++mi)
            #pragma unroll
            for (int ni = 0; ni < 4; ++ni)
                acc[mi][ni] = __builtin_amdgcn_wmma_f32_16x16x32_f16(
                    false, af[mi].v, false, bf[ni].v, (short)0, acc[mi][ni], false, false);
    }

    // ---- epilogue: bias / accumulate / relu / f32 + f16 stores ----
    #pragma unroll
    for (int mi = 0; mi < 2; ++mi) {
        #pragma unroll
        for (int ni = 0; ni < 4; ++ni) {
            const int ncol = n0 + wn * 64 + ni * 16 + l16;
            const float bv = bias ? bias[ncol] * bias_scale : 0.0f;
            #pragma unroll
            for (int r = 0; r < 8; ++r) {
                const int mrow = m0 + wm * 32 + mi * 16 + half16 * 8 + r;
                if (mrow < M) {
                    const size_t o = (size_t)mrow * Nc + ncol;
                    float v = acc[mi][ni][r] + bv;
                    if (flags & GF_ACC)  v += C[o];
                    if (flags & GF_RELU) v = fmaxf(v, 0.0f);
                    if (C)  C[o]  = v;
                    if (Ch) Ch[o] = (_Float16)v;
                }
            }
        }
    }
}

// ---------------------------------------------------------------------------
// Elementwise / graph kernels
// ---------------------------------------------------------------------------
__global__ void copy_f32(const float* __restrict__ a, float* __restrict__ b, int n) {
    int i = blockIdx.x * blockDim.x + threadIdx.x;
    if (i < n) b[i] = a[i];
}
__global__ void f32_to_f16(const float* __restrict__ a, _Float16* __restrict__ b, int n) {
    int i = blockIdx.x * blockDim.x + threadIdx.x;
    if (i < n) b[i] = (_Float16)a[i];
}
// weight convert + transpose: BT[n*K + k] = (f16) B[k*Nc + n]
__global__ void f32_to_f16_T(const float* __restrict__ B, _Float16* __restrict__ BT,
                             int K, int Nc) {
    int i = blockIdx.x * blockDim.x + threadIdx.x;
    if (i >= K * Nc) return;
    const int k = i / Nc, n = i - k * Nc;
    BT[(size_t)n * K + k] = (_Float16)B[i];
}
__global__ void zero_f16(_Float16* a, int n) {
    int i = blockIdx.x * blockDim.x + threadIdx.x;
    if (i < n) a[i] = (_Float16)0.0f;
}
__global__ void zero_f32(float* a, int n) {
    int i = blockIdx.x * blockDim.x + threadIdx.x;
    if (i < n) a[i] = 0.0f;
}

// agg[dst] += x[src], vectorized 4 channels / thread, f32 atomics
__global__ void scatter_edges(const float* __restrict__ x, const int* __restrict__ src,
                              const int* __restrict__ dst, float* __restrict__ agg,
                              int E, int F)
{
    const int per = F >> 2;
    int idx = blockIdx.x * blockDim.x + threadIdx.x;
    if (idx >= E * per) return;
    const int e = idx / per;
    const int c = (idx - e * per) << 2;
    const int s = src[e], d = dst[e];
    const float4 v = *(const float4*)(x + (size_t)s * F + c);
    float* ap = agg + (size_t)d * F + c;
    atomicAdd(ap + 0, v.x); atomicAdd(ap + 1, v.y);
    atomicAdd(ap + 2, v.z); atomicAdd(ap + 3, v.w);
}

// residual cascade + pooling (+ f16 copies of the post-residual seq for LSTM)
__global__ void residual_pool(float* x1, float* x2, float* x3, float* x4,
                              _Float16* xh1, _Float16* xh2, _Float16* xh3, _Float16* xh4,
                              float* comb, int n)
{
    int i = blockIdx.x * blockDim.x + threadIdx.x;
    if (i >= n) return;
    const float a = x1[i];
    const float b = x2[i] + a;  x2[i] = b;
    const float c = x3[i] + b;  x3[i] = c;
    const float d = x4[i] + c;  x4[i] = d;
    const float s = a + b + c + d;                     // == x_sum
    const float avg = 0.4f * s;                        // (s + a+b+c+d)/5
    const float mx  = fmaxf(s, fmaxf(fmaxf(a, b), fmaxf(c, d)));
    comb[i] = avg + mx;                                // x_avg + x_max
    xh1[i] = (_Float16)a; xh2[i] = (_Float16)b;
    xh3[i] = (_Float16)c; xh4[i] = (_Float16)d;
}

// LSTM pointwise: gates -> (c,h); optional running mean into comb
__global__ void lstm_gate(const float* __restrict__ g, float* __restrict__ c,
                          _Float16* __restrict__ h_out, float* __restrict__ comb,
                          float combScale, int nh)
{
    int i = blockIdx.x * blockDim.x + threadIdx.x;
    if (i >= nh) return;
    const int node = i >> 8;          // H = 256
    const int ch   = i & 255;
    const float* gr = g + (size_t)node * G4;
    const float ig = gr[ch], fg = gr[256 + ch], gg = gr[512 + ch], og = gr[768 + ch];
    const float si = 1.0f / (1.0f + __expf(-ig));
    const float sf = 1.0f / (1.0f + __expf(-fg));
    const float so = 1.0f / (1.0f + __expf(-og));
    const float cv = sf * c[i] + si * tanhf(gg);
    c[i] = cv;
    const float hv = so * tanhf(cv);
    h_out[i] = (_Float16)hv;
    if (comb) comb[i] += combScale * hv;
}

// ---------------------------------------------------------------------------
// Host orchestration
// ---------------------------------------------------------------------------
static inline int ceil_div(int a, int b) { return (a + b - 1) / b; }

extern "C" void kernel_launch(void* const* d_in, const int* in_sizes, int n_in,
                              void* d_out, int out_size, void* d_ws, size_t ws_size,
                              hipStream_t stream)
{
    (void)in_sizes; (void)n_in; (void)out_size; (void)ws_size;

    const float* x   = (const float*)d_in[0];
    const int*   ei  = (const int*)d_in[1];
    const int*   src = ei;
    const int*   dst = ei + EE;

    const float* w1[4]; const float* b1[4]; const float* w2[4]; const float* b2[4];
    for (int l = 0; l < 4; ++l) {
        w1[l] = (const float*)d_in[2 + 4 * l];
        b1[l] = (const float*)d_in[3 + 4 * l];
        w2[l] = (const float*)d_in[4 + 4 * l];
        b2[l] = (const float*)d_in[5 + 4 * l];
    }
    const float* wih[2] = {(const float*)d_in[18], (const float*)d_in[21]};
    const float* whh[2] = {(const float*)d_in[19], (const float*)d_in[22]};
    const float* lb[2]  = {(const float*)d_in[20], (const float*)d_in[23]};
    const float* mlp_w  = (const float*)d_in[24];
    const float* mlp_b  = (const float*)d_in[25];

    // ---- workspace carving ----
    char* wsp = (char*)d_ws;
    size_t off = 0;
    auto take = [&](size_t bytes) -> char* {
        char* p = wsp + off;
        off = (off + bytes + 255) & ~(size_t)255;
        return p;
    };
    float*    g      = (float*)take((size_t)NN * G4 * 4);        // gates / conv scratch
    float*    xs[4];  for (int i = 0; i < 4; ++i) xs[i]  = (float*)take((size_t)NN * HH * 4);
    _Float16* xh[4];  for (int i = 0; i < 4; ++i) xh[i]  = (_Float16*)take((size_t)NN * HH * 2);
    _Float16* hl0[4]; for (int i = 0; i < 4; ++i) hl0[i] = (_Float16*)take((size_t)NN * HH * 2);
    _Float16* h1h    = (_Float16*)take((size_t)NN * HH * 2);
    _Float16* hz     = (_Float16*)take((size_t)NN * HH * 2);
    float*    cbuf   = (float*)take((size_t)NN * HH * 4);
    float*    comb   = (float*)take((size_t)NN * HH * 4);
    _Float16* comb_h = (_Float16*)take((size_t)NN * HH * 2);
    // f16 transposed weights
    _Float16* w1h[4]; _Float16* w2h[4];
    w1h[0] = (_Float16*)take((size_t)F_IN * HH * 2);
    for (int l = 1; l < 4; ++l) w1h[l] = (_Float16*)take((size_t)HH * HH * 2);
    for (int l = 0; l < 4; ++l) w2h[l] = (_Float16*)take((size_t)HH * HH * 2);
    _Float16* wihh[2]; _Float16* whhh[2];
    for (int l = 0; l < 2; ++l) { wihh[l] = (_Float16*)take((size_t)HH * G4 * 2);
                                  whhh[l] = (_Float16*)take((size_t)HH * G4 * 2); }
    _Float16* mlpwh = (_Float16*)take((size_t)HH * HH * 2);
    // conv scratch aliases inside g (gates unused during conv phase)
    float*    aggF = g;
    _Float16* aggH = (_Float16*)((char*)g + (size_t)NN * F_IN * 4);
    _Float16* zH   = (_Float16*)((char*)g + (size_t)NN * F_IN * 4 + (size_t)NN * F_IN * 2);

    const int TB = 256;
    auto cvt = [&](const float* a, _Float16* b, int n) {
        f32_to_f16<<<ceil_div(n, TB), TB, 0, stream>>>(a, b, n);
    };
    auto cvtT = [&](const float* B, _Float16* BT, int K, int Nc) {
        f32_to_f16_T<<<ceil_div(K * Nc, TB), TB, 0, stream>>>(B, BT, K, Nc);
    };
    auto gemm = [&](const _Float16* A, const _Float16* BT, const float* bias, float bscale,
                    float* C, _Float16* Ch, int M, int K, int Nc, int flags) {
        dim3 grid(Nc / TILE_N, ceil_div(M, TILE_M));
        wmma_gemm<<<grid, dim3(256), 0, stream>>>(A, BT, bias, bscale, C, Ch, M, K, Nc, flags);
    };

    // ---- convert + transpose weights to f16 ----
    cvtT(w1[0], w1h[0], F_IN, HH);
    for (int l = 1; l < 4; ++l) cvtT(w1[l], w1h[l], HH, HH);
    for (int l = 0; l < 4; ++l) cvtT(w2[l], w2h[l], HH, HH);
    for (int l = 0; l < 2; ++l) { cvtT(wih[l], wihh[l], HH, G4); cvtT(whh[l], whhh[l], HH, G4); }
    cvtT(mlp_w, mlpwh, HH, HH);

    // ---- 4 GIN conv layers ----
    const float* hin = x;
    int Fin = F_IN;
    for (int l = 0; l < 4; ++l) {
        const int nF = NN * Fin;
        copy_f32<<<ceil_div(nF, TB), TB, 0, stream>>>(hin, aggF, nF);      // self term
        scatter_edges<<<ceil_div(EE * (Fin >> 2), TB), TB, 0, stream>>>(hin, src, dst, aggF, EE, Fin);
        cvt(aggF, aggH, nF);
        gemm(aggH, w1h[l], b1[l], 1.0f, nullptr, zH, NN, Fin, HH, GF_RELU);      // z (f16)
        gemm(zH,   w2h[l], b2[l], 1.0f, xs[l],  nullptr, NN, HH,  HH, GF_RELU);  // x_l (f32)
        hin = xs[l];
        Fin = HH;
    }

    // ---- residual cascade + avg/max pooling ----
    residual_pool<<<ceil_div(NN * HH, TB), TB, 0, stream>>>(
        xs[0], xs[1], xs[2], xs[3], xh[0], xh[1], xh[2], xh[3], comb, NN * HH);

    // ---- 2-layer LSTM over T=4 ----
    zero_f16<<<ceil_div(NN * HH, TB), TB, 0, stream>>>(hz, NN * HH);
    zero_f32<<<ceil_div(NN * HH, TB), TB, 0, stream>>>(cbuf, NN * HH);
    for (int t = 0; t < 4; ++t) {   // layer 0: h_t stored per-step for layer 1
        const _Float16* hprev = (t == 0) ? hz : hl0[t - 1];
        gemm(xh[t], wihh[0], lb[0], 1.0f, g, nullptr, NN, HH, G4, 0);
        gemm(hprev, whhh[0], nullptr, 0.0f, g, nullptr, NN, HH, G4, GF_ACC);
        lstm_gate<<<ceil_div(NN * HH, TB), TB, 0, stream>>>(g, cbuf, hl0[t], nullptr, 0.0f, NN * HH);
    }
    zero_f16<<<ceil_div(NN * HH, TB), TB, 0, stream>>>(h1h, NN * HH);
    zero_f32<<<ceil_div(NN * HH, TB), TB, 0, stream>>>(cbuf, NN * HH);
    for (int t = 0; t < 4; ++t) {   // layer 1: mean(h) accumulated into comb
        const _Float16* hprev = (t == 0) ? hz : h1h;
        gemm(hl0[t], wihh[1], lb[1], 1.0f, g, nullptr, NN, HH, G4, 0);
        gemm(hprev,  whhh[1], nullptr, 0.0f, g, nullptr, NN, HH, G4, GF_ACC);
        lstm_gate<<<ceil_div(NN * HH, TB), TB, 0, stream>>>(g, cbuf, h1h, comb, 0.25f, NN * HH);
    }

    // ---- final: relu((x_avg + x_max + x_lstm) @ mlp_w + 3*mlp_b) ----
    cvt(comb, comb_h, NN * HH);
    gemm(comb_h, mlpwh, mlp_b, 3.0f, (float*)d_out, nullptr, NN, HH, HH, GF_RELU);
}
